// TransformerBlock_44770739094118
// MI455X (gfx1250) — matrix operand order repaired and checked by reading, not compiled, not run on hardware
//
#include <hip/hip_runtime.h>
#include <stdint.h>

// ---------------------------------------------------------------------------
// CDNA5 (gfx1250) transformer block: bf16 WMMA GEMMs + flash attention.
// wave32 everywhere; v_wmma_f32_16x16x32_bf16 is the compute workhorse.
// GEMM tile staging uses GLOBAL_LOAD_ASYNC_TO_LDS_B128 (ASYNCcnt) so the
// global->LDS copy bypasses VGPRs and overlaps the WMMA block.
// ---------------------------------------------------------------------------

typedef __attribute__((ext_vector_type(16))) __bf16 v16bf;
typedef __attribute__((ext_vector_type(8)))  float  v8f;

struct B8 { unsigned u[8]; };

__device__ __forceinline__ v16bf mk_frag(const B8& x) {
  return __builtin_bit_cast(v16bf, x);
}

__device__ __forceinline__ unsigned short f2bf(float f) {
  union { float f; unsigned u; } cv; cv.f = f;
  unsigned u = cv.u;
  u += 0x7FFFu + ((u >> 16) & 1u);   // round-to-nearest-even
  return (unsigned short)(u >> 16);
}

__device__ __forceinline__ v8f wmma_bf16(v16bf a, v16bf b, v8f c) {
  // D = A(16x32 bf16) * B(32x16 bf16) + C(16x16 f32)
  return __builtin_amdgcn_wmma_f32_16x16x32_bf16(
      /*neg_a=*/false, a, /*neg_b=*/false, b,
      /*c_mod=*/(short)0, c, /*reuse_a=*/false, /*reuse_b=*/false);
}

// Async DMA: 16B per lane, global -> LDS, tracked by ASYNCcnt.
// lds_off = LDS byte address (low 32 bits of the flat shared pointer).
__device__ __forceinline__ void async_ld_b128(void* lds_ptr, const void* gaddr) {
  unsigned lds_off = (unsigned)(size_t)lds_ptr;
  asm volatile("global_load_async_to_lds_b128 %0, %1, off"
               :: "v"(lds_off), "v"(gaddr)
               : "memory");
}
__device__ __forceinline__ void wait_async0() {
  asm volatile("s_wait_asynccnt 0" ::: "memory");
}

// ---------------------------------------------------------------------------
// Weight convert + transpose: src f32 [K][N] row-major -> dst bf16 [N][K]
// ---------------------------------------------------------------------------
__global__ __launch_bounds__(256) void cvtT_kernel(
    const float* __restrict__ src, unsigned short* __restrict__ dst,
    int K, int N) {
  long i = (long)blockIdx.x * 256 + threadIdx.x;
  long total = (long)K * N;
  if (i >= total) return;
  int k = (int)(i / N);
  int n = (int)(i % N);
  dst[(size_t)n * K + k] = f2bf(src[i]);
}

// ---------------------------------------------------------------------------
// LayerNorm over C=1024, one row per 256-thread block; bf16 output.
// ---------------------------------------------------------------------------
__global__ __launch_bounds__(256) void ln_kernel(
    const float* __restrict__ x, const float* __restrict__ g,
    const float* __restrict__ b, unsigned short* __restrict__ out, int C) {
  const int row = blockIdx.x;
  const float* xr = x + (size_t)row * C;
  __shared__ float red[256];
  float s = 0.f, s2 = 0.f;
  for (int i = threadIdx.x; i < C; i += 256) {
    float v = xr[i]; s += v; s2 += v * v;
  }
  red[threadIdx.x] = s; __syncthreads();
  for (int off = 128; off; off >>= 1) {
    if (threadIdx.x < off) red[threadIdx.x] += red[threadIdx.x + off];
    __syncthreads();
  }
  float mean = red[0] / C; __syncthreads();
  red[threadIdx.x] = s2; __syncthreads();
  for (int off = 128; off; off >>= 1) {
    if (threadIdx.x < off) red[threadIdx.x] += red[threadIdx.x + off];
    __syncthreads();
  }
  float var = red[0] / C - mean * mean;
  float inv = rsqrtf(var + 1e-5f);
  for (int i = threadIdx.x; i < C; i += 256) {
    float v = (xr[i] - mean) * inv * g[i] + b[i];
    out[(size_t)row * C + i] = f2bf(v);
  }
}

// ---------------------------------------------------------------------------
// Tiled bf16 GEMM: C[M][N] = A[M][K] @ W[K][N] (+epilogue).
// A row-major bf16, BT = W^T row-major bf16 [N][K].
// Block: 256 threads = 8 waves (2 along M x 4 along N), tile 64x256, BK=32.
// Each wave: 32x64 = 2x4 frags -> 8 v_wmma per k-step.
// Double-buffered LDS; the next tile is DMA'd with async-to-LDS loads issued
// at the top of the (straight-line) body, overlapped with the 8 WMMA, then
// fenced with s_wait_asynccnt 0 + one barrier per k-step.
// ---------------------------------------------------------------------------
#define BM 64
#define BN 256
#define BK 32
#define LDT 40  // padded LDS row stride in bf16 elems (80B, 16B-aligned)

#define EPI_QKV  0
#define EPI_PROJ 1
#define EPI_FC   2
#define EPI_FC2  3

__global__ __launch_bounds__(256, 1) void gemm_bf16_kernel(
    const unsigned short* __restrict__ A,    // [M][K] bf16
    const unsigned short* __restrict__ BT,   // [N][K] bf16
    const float* __restrict__ bias,          // [N] f32
    int M, int N, int K, int epi,
    unsigned short* __restrict__ qb,         // EPI_QKV: [BH][T][64]
    unsigned short* __restrict__ kb,         // EPI_QKV: [BH][T][64]
    unsigned short* __restrict__ vb,         // EPI_QKV: [BH][64][T]
    const float* __restrict__ resid,         // EPI_PROJ/FC2: [M][N]
    float* __restrict__ out_f32,             // EPI_PROJ/FC2: [M][N]
    unsigned short* __restrict__ out_bf16) { // EPI_FC: [M][N]
  __shared__ __align__(16) unsigned short As[2][BM * LDT];
  __shared__ __align__(16) unsigned short Bs[2][BN * LDT];

  const int tid   = threadIdx.x;
  const int lane  = tid & 31;
  const int wid   = tid >> 5;
  const int waveM = wid >> 2;   // 0..1 (32 rows each)
  const int waveN = wid & 3;    // 0..3 (64 cols each)
  const int hf    = lane >> 4;  // lane half (0/1)
  const int nl    = lane & 15;
  const int row0  = blockIdx.y * BM;
  const int col0  = blockIdx.x * BN;

  // staging thread -> tile coords (precomputed once)
  const int ar = tid >> 2;            // A row 0..63
  const int ac = (tid & 3) * 8;       // A col {0,8,16,24}
  int br[4], bc[4];
  #pragma unroll
  for (int it = 0; it < 4; ++it) {
    int idx = tid + it * 256;         // 0..1023
    br[it] = idx >> 2;                // B row 0..255
    bc[it] = (idx & 3) * 8;
  }
  const unsigned short* Abase = A + (size_t)(row0 + ar) * K + ac;
  const unsigned short* Bb    = BT + (size_t)col0 * K;

  v8f zero = {};
  v8f acc[2][4];
  #pragma unroll
  for (int i = 0; i < 2; ++i)
    #pragma unroll
    for (int j = 0; j < 4; ++j) acc[i][j] = zero;

  // ---- preload tile 0 (async DMA) ----
  async_ld_b128(&As[0][ar * LDT + ac], Abase);
  #pragma unroll
  for (int it = 0; it < 4; ++it)
    async_ld_b128(&Bs[0][br[it] * LDT + bc[it]],
                  Bb + (size_t)br[it] * K + bc[it]);
  wait_async0();
  __syncthreads();

  int cur = 0;
  for (int k0 = 0; k0 < K; k0 += BK) {
    // wrapped next-tile index: last iteration harmlessly reloads tile 0
    const int k1 = (k0 + BK < K) ? (k0 + BK) : 0;
    const int nxt = cur ^ 1;

    // ---- issue next-tile async DMA (overlaps the WMMA block) ----
    async_ld_b128(&As[nxt][ar * LDT + ac], Abase + k1);
    #pragma unroll
    for (int it = 0; it < 4; ++it)
      async_ld_b128(&Bs[nxt][br[it] * LDT + bc[it]],
                    Bb + (size_t)br[it] * K + bc[it] + k1);

    // ---- fragments from current buffer + 8 WMMA ----
    const unsigned* As32 = (const unsigned*)As[cur];
    const unsigned* Bs32 = (const unsigned*)Bs[cur];
    v16bf afrag[2];
    #pragma unroll
    for (int fm = 0; fm < 2; ++fm) {
      int r = waveM * 32 + fm * 16 + nl;  // A row (M), lane = row
      B8 t;
      #pragma unroll
      for (int vi = 0; vi < 8; ++vi) {
        int kk = ((vi >> 2) << 4) + (hf << 3) + ((vi & 3) << 1);  // A layout
        t.u[vi] = As32[(r * LDT + kk) >> 1];
      }
      afrag[fm] = mk_frag(t);
    }
    #pragma unroll
    for (int fn = 0; fn < 4; ++fn) {
      int r = waveN * 64 + fn * 16 + nl;  // B col (N), lane = col
      B8 t;
      #pragma unroll
      for (int vi = 0; vi < 8; ++vi) {
        int kk = ((vi >> 2) << 4) + (hf << 3) + ((vi & 3) << 1);  // B layout
        t.u[vi] = Bs32[(r * LDT + kk) >> 1];
      }
      v16bf bf = mk_frag(t);
      acc[0][fn] = wmma_bf16(afrag[0], bf, acc[0][fn]);
      acc[1][fn] = wmma_bf16(afrag[1], bf, acc[1][fn]);
    }

    // ---- fence DMA + publish ----
    wait_async0();
    __syncthreads();
    cur ^= 1;
  }

  // ---- epilogue: C/D layout: m = hf*8 + vi, n = nl ----
  #pragma unroll
  for (int fm = 0; fm < 2; ++fm) {
    #pragma unroll
    for (int fn = 0; fn < 4; ++fn) {
      int rbase = row0 + waveM * 32 + fm * 16 + hf * 8;
      int col   = col0 + waveN * 64 + fn * 16 + nl;
      float bv = bias[col];
      #pragma unroll
      for (int vi = 0; vi < 8; ++vi) {
        int row = rbase + vi;
        float val = acc[fm][fn][vi] + bv;
        if (epi == EPI_QKV) {
          int which = col >> 10;            // 0=q 1=k 2=v
          int h = (col >> 6) & 15;
          int d = col & 63;
          int bidx = row >> 11;             // T = 2048
          int t = row & 2047;
          size_t bh = (size_t)bidx * 16 + h;
          if (which == 0)      qb[(bh * 2048 + t) * 64 + d] = f2bf(val * 0.125f);
          else if (which == 1) kb[(bh * 2048 + t) * 64 + d] = f2bf(val);
          else                 vb[(bh * 64 + d) * 2048 + t] = f2bf(val);
        } else if (epi == EPI_PROJ) {
          out_f32[(size_t)row * N + col] = val + resid[(size_t)row * N + col];
        } else if (epi == EPI_FC) {
          // gelu_tanh(u) = u * 0.5*(1+tanh(z)) = u / (1 + exp(-2z))
          float u = val;
          float z = 0.7978845608028654f * (u + 0.044715f * u * u * u);
          float gl = u / (1.0f + __expf(-2.0f * z));
          out_bf16[(size_t)row * N + col] = f2bf(gl);
        } else {  // EPI_FC2
          out_f32[(size_t)row * N + col] = val + resid[(size_t)row * N + col];
        }
      }
    }
  }
}

// ---------------------------------------------------------------------------
// Flash attention: one wave per 16-row q tile; 32 keys per iteration.
// q: [BH][T][64] (pre-scaled by 1/sqrt(64)); k: [BH][T][64]; vT: [BH][64][T].
// Output y bf16 [B*T][1024].
// ---------------------------------------------------------------------------
__global__ __launch_bounds__(256, 1) void attn_kernel(
    const unsigned short* __restrict__ q,
    const unsigned short* __restrict__ kmat,
    const unsigned short* __restrict__ vT,
    unsigned short* __restrict__ y, int T) {
  __shared__ __align__(16) unsigned short ptile[8][16 * 32];  // per-wave P tile

  const int lane = threadIdx.x & 31;
  const int wid  = threadIdx.x >> 5;
  const int hf   = lane >> 4;
  const int nl   = lane & 15;
  const int bh   = blockIdx.y;           // 0..63
  const int bidx = bh >> 4;
  const int h    = bh & 15;
  const int qtile = blockIdx.x * 8 + wid;
  const int q0 = qtile * 16;

  const unsigned short* qp = q    + (size_t)bh * T * 64;
  const unsigned short* kp = kmat + (size_t)bh * T * 64;
  const unsigned short* vp = vT   + (size_t)bh * 64 * T;

  // Q fragments (A layout), head-dim halves 0..31 and 32..63
  v16bf qa[2];
  {
    const unsigned* q32 = (const unsigned*)(qp + (size_t)(q0 + nl) * 64);
    #pragma unroll
    for (int f = 0; f < 2; ++f) {
      B8 t;
      #pragma unroll
      for (int vi = 0; vi < 8; ++vi) {
        int kk = f * 32 + ((vi >> 2) << 4) + (hf << 3) + ((vi & 3) << 1);
        t.u[vi] = q32[kk >> 1];
      }
      qa[f] = mk_frag(t);
    }
  }

  v8f zero = {};
  v8f o[4]; o[0] = zero; o[1] = zero; o[2] = zero; o[3] = zero;
  float mrun[8], lrun[8];
  #pragma unroll
  for (int vi = 0; vi < 8; ++vi) { mrun[vi] = -1e30f; lrun[vi] = 0.f; }

  const int numk = q0 + 16;  // causal: keys [0, numk)
  for (int kbase = 0; kbase < numk; kbase += 32) {
    // ---- S = Q @ K^T for 32 keys (two 16-key n-tiles) ----
    v8f s[2];
    #pragma unroll
    for (int nt = 0; nt < 2; ++nt) {
      int key = kbase + nt * 16 + nl;
      const unsigned* k32 = (const unsigned*)(kp + (size_t)key * 64);
      v8f sacc = zero;
      #pragma unroll
      for (int f = 0; f < 2; ++f) {
        B8 t;
        #pragma unroll
        for (int vi = 0; vi < 8; ++vi) {
          int kk = f * 32 + ((vi >> 2) << 4) + (hf << 3) + ((vi & 3) << 1);  // B layout
          t.u[vi] = k32[kk >> 1];
        }
        sacc = wmma_bf16(qa[f], mk_frag(t), sacc);
      }
      s[nt] = sacc;
    }

    // ---- causal mask + online softmax ----
    float pmax[8];
    #pragma unroll
    for (int vi = 0; vi < 8; ++vi) {
      int qrow = q0 + hf * 8 + vi;
      if (kbase + nl > qrow)      s[0][vi] = -1e30f;
      if (kbase + 16 + nl > qrow) s[1][vi] = -1e30f;
      pmax[vi] = fmaxf(s[0][vi], s[1][vi]);
    }
    #pragma unroll
    for (int off = 1; off < 16; off <<= 1)
      #pragma unroll
      for (int vi = 0; vi < 8; ++vi)
        pmax[vi] = fmaxf(pmax[vi], __shfl_xor(pmax[vi], off, 32));

    float psum[8];
    #pragma unroll
    for (int vi = 0; vi < 8; ++vi) {
      float mnew = fmaxf(mrun[vi], pmax[vi]);
      float alpha = __expf(mrun[vi] - mnew);
      s[0][vi] = __expf(s[0][vi] - mnew);
      s[1][vi] = __expf(s[1][vi] - mnew);
      psum[vi] = s[0][vi] + s[1][vi];
      lrun[vi] *= alpha;
      mrun[vi] = mnew;
      o[0][vi] *= alpha; o[1][vi] *= alpha; o[2][vi] *= alpha; o[3][vi] *= alpha;
    }
    #pragma unroll
    for (int off = 1; off < 16; off <<= 1)
      #pragma unroll
      for (int vi = 0; vi < 8; ++vi)
        psum[vi] += __shfl_xor(psum[vi], off, 32);
    #pragma unroll
    for (int vi = 0; vi < 8; ++vi) lrun[vi] += psum[vi];

    // ---- P (C layout) -> LDS -> A fragment (lane transpose) ----
    unsigned short* pt = ptile[wid];
    #pragma unroll
    for (int vi = 0; vi < 8; ++vi) {
      int m = hf * 8 + vi;
      pt[m * 32 + nl]      = f2bf(s[0][vi]);
      pt[m * 32 + 16 + nl] = f2bf(s[1][vi]);
    }
    v16bf pa;
    {
      const unsigned* p32 = (const unsigned*)(pt + nl * 32);
      B8 t;
      #pragma unroll
      for (int vi = 0; vi < 8; ++vi) {
        int kk = ((vi >> 2) << 4) + (hf << 3) + ((vi & 3) << 1);
        t.u[vi] = p32[kk >> 1];
      }
      pa = mk_frag(t);
    }

    // ---- O += P @ V (4 head-dim n-tiles) ----
    #pragma unroll
    for (int dt = 0; dt < 4; ++dt) {
      const unsigned* v32 =
          (const unsigned*)(vp + (size_t)(dt * 16 + nl) * T + kbase);
      B8 t;
      #pragma unroll
      for (int vi = 0; vi < 8; ++vi) {
        int kk = ((vi >> 2) << 4) + (hf << 3) + ((vi & 3) << 1);
        t.u[vi] = v32[kk >> 1];
      }
      o[dt] = wmma_bf16(pa, mk_frag(t), o[dt]);
    }
  }

  // ---- normalize + store y bf16 [B*T][1024] ----
  #pragma unroll
  for (int dt = 0; dt < 4; ++dt) {
    #pragma unroll
    for (int vi = 0; vi < 8; ++vi) {
      int m = hf * 8 + vi;
      int row = bidx * T + q0 + m;
      float val = o[dt][vi] / lrun[vi];
      y[(size_t)row * 1024 + h * 64 + dt * 16 + nl] = f2bf(val);
    }
  }
}

// ---------------------------------------------------------------------------
// Host launcher
// ---------------------------------------------------------------------------
extern "C" void kernel_launch(void* const* d_in, const int* in_sizes, int n_in,
                              void* d_out, int out_size, void* d_ws,
                              size_t ws_size, hipStream_t stream) {
  (void)in_sizes; (void)n_in; (void)out_size; (void)ws_size;
  const int Bb = 4, T = 2048, C = 1024, FF = 4096;
  const int M = Bb * T;  // 8192

  const float* x      = (const float*)d_in[0];
  const float* ln1_g  = (const float*)d_in[1];
  const float* ln1_b  = (const float*)d_in[2];
  const float* w_attn = (const float*)d_in[3];
  const float* b_attn = (const float*)d_in[4];
  const float* w_proj = (const float*)d_in[5];
  const float* b_proj = (const float*)d_in[6];
  const float* ln2_g  = (const float*)d_in[7];
  const float* ln2_b  = (const float*)d_in[8];
  const float* w_fc   = (const float*)d_in[9];
  const float* b_fc   = (const float*)d_in[10];
  const float* w_fc2  = (const float*)d_in[11];
  const float* b_fc2  = (const float*)d_in[12];
  float* out = (float*)d_out;

  char* ws = (char*)d_ws;
  size_t off = 0;
  auto alloc = [&](size_t bytes) -> char* {
    char* p = ws + off;
    off = (off + bytes + 255) & ~(size_t)255;
    return p;
  };
  unsigned short* wTattn = (unsigned short*)alloc((size_t)3 * C * C * 2);
  unsigned short* wTproj = (unsigned short*)alloc((size_t)C * C * 2);
  unsigned short* wTfc   = (unsigned short*)alloc((size_t)FF * C * 2);
  unsigned short* wTfc2  = (unsigned short*)alloc((size_t)C * FF * 2);
  unsigned short* h1     = (unsigned short*)alloc((size_t)M * C * 2);
  unsigned short* qb     = (unsigned short*)alloc((size_t)M * C * 2);
  unsigned short* kbuf   = (unsigned short*)alloc((size_t)M * C * 2);
  unsigned short* vbuf   = (unsigned short*)alloc((size_t)M * C * 2);
  unsigned short* ybuf   = (unsigned short*)alloc((size_t)M * C * 2);
  float*          x2     = (float*)alloc((size_t)M * C * 4);
  unsigned short* h2     = (unsigned short*)alloc((size_t)M * C * 2);
  unsigned short* hfc    = (unsigned short*)alloc((size_t)M * FF * 2);

  // 1) weights -> bf16 transposed [N][K]
  {
    long t1 = (long)C * 3 * C;
    cvtT_kernel<<<(t1 + 255) / 256, 256, 0, stream>>>(w_attn, wTattn, C, 3 * C);
    long t2 = (long)C * C;
    cvtT_kernel<<<(t2 + 255) / 256, 256, 0, stream>>>(w_proj, wTproj, C, C);
    long t3 = (long)C * FF;
    cvtT_kernel<<<(t3 + 255) / 256, 256, 0, stream>>>(w_fc, wTfc, C, FF);
    long t4 = (long)FF * C;
    cvtT_kernel<<<(t4 + 255) / 256, 256, 0, stream>>>(w_fc2, wTfc2, FF, C);
  }

  // 2) LN1 -> h1 bf16
  ln_kernel<<<M, 256, 0, stream>>>(x, ln1_g, ln1_b, h1, C);

  // 3) QKV GEMM: [M,C] @ [C,3C], scatter to q/k/vT bf16
  gemm_bf16_kernel<<<dim3(3 * C / BN, M / BM), 256, 0, stream>>>(
      h1, wTattn, b_attn, M, 3 * C, C, EPI_QKV,
      qb, kbuf, vbuf, nullptr, nullptr, nullptr);

  // 4) flash attention -> y bf16
  attn_kernel<<<dim3(T / 16 / 8, Bb * 16), 256, 0, stream>>>(qb, kbuf, vbuf,
                                                             ybuf, T);

  // 5) proj GEMM + residual -> x2 f32
  gemm_bf16_kernel<<<dim3(C / BN, M / BM), 256, 0, stream>>>(
      ybuf, wTproj, b_proj, M, C, C, EPI_PROJ,
      nullptr, nullptr, nullptr, x, x2, nullptr);

  // 6) LN2 -> h2 bf16
  ln_kernel<<<M, 256, 0, stream>>>(x2, ln2_g, ln2_b, h2, C);

  // 7) FC GEMM + GELU -> hfc bf16
  gemm_bf16_kernel<<<dim3(FF / BN, M / BM), 256, 0, stream>>>(
      h2, wTfc, b_fc, M, FF, C, EPI_FC,
      nullptr, nullptr, nullptr, nullptr, nullptr, hfc);

  // 8) FC2 GEMM + residual -> d_out f32
  gemm_bf16_kernel<<<dim3(C / BN, M / BM), 256, 0, stream>>>(
      hfc, wTfc2, b_fc2, M, C, FF, EPI_FC2,
      nullptr, nullptr, nullptr, x2, out, nullptr);
}
